// TransformerBlock_82901458747704
// MI455X (gfx1250) — matrix operand order repaired and checked by reading, not compiled, not run on hardware
//
#include <hip/hip_runtime.h>
#include <hip/hip_bf16.h>
#include <math.h>

typedef __bf16 bf16;
typedef __attribute__((ext_vector_type(16))) __bf16 v16bf;
typedef __attribute__((ext_vector_type(8)))  float   v8f;

// ---------------------------------------------------------------------------
// helpers
// ---------------------------------------------------------------------------
__device__ __forceinline__ float block_reduce(float v, float* red, int tid, bool do_max) {
  red[tid] = v;
  __syncthreads();
  for (int s = 128; s > 0; s >>= 1) {
    if (tid < s) red[tid] = do_max ? fmaxf(red[tid], red[tid + s]) : (red[tid] + red[tid + s]);
    __syncthreads();
  }
  float r = red[0];
  __syncthreads();
  return r;
}

// lambda scalar: exp(dot(q1,k1)) - exp(dot(q2,k2)) + lambda_init
__global__ __launch_bounds__(256)
void lam_kernel(const float* __restrict__ q1, const float* __restrict__ k1,
                const float* __restrict__ q2, const float* __restrict__ k2,
                float* __restrict__ out, int n, float lambda_init) {
  __shared__ float red[256];
  int tid = threadIdx.x;
  float a = 0.f, b = 0.f;
  for (int i = tid; i < n; i += 256) { a += q1[i] * k1[i]; b += q2[i] * k2[i]; }
  float s1 = block_reduce(a, red, tid, false);
  float s2 = block_reduce(b, red, tid, false);
  if (tid == 0) out[0] = expf(s1) - expf(s2) + lambda_init;
}

// per-(b,c) inverse RMS over the seq axis of X[B][S][C]
__global__ __launch_bounds__(256)
void colstats(const float* __restrict__ X, float* __restrict__ inv,
              int S, int Cc, float eps) {
  const int b = blockIdx.y;
  const int c0 = blockIdx.x * 32;
  const int tid = threadIdx.x;
  const int cc = tid & 31;
  const int sg = tid >> 5;
  float acc = 0.f;
  for (int s = sg; s < S; s += 8) {
    float v = X[((long)b * S + s) * Cc + c0 + cc];
    acc += v * v;
  }
  __shared__ float red[256];
  red[tid] = acc;
  __syncthreads();
  if (tid < 32) {
    float t = red[tid];
    for (int j = 1; j < 8; ++j) t += red[j * 32 + tid];
    inv[(long)b * Cc + c0 + tid] = rsqrtf(t / (float)S + eps);
  }
}

// per-row inverse RMS of contiguous rows of length L
__global__ __launch_bounds__(256)
void rowstats(const float* __restrict__ X, float* __restrict__ inv, int L, float eps) {
  const long row = blockIdx.x;
  const int tid = threadIdx.x;
  float acc = 0.f;
  for (int i = tid; i < L; i += 256) {
    float v = X[row * L + i];
    acc += v * v;
  }
  __shared__ float red[256];
  float s = block_reduce(acc, red, tid, false);
  if (tid == 0) inv[row] = rsqrtf(s / (float)L + eps);
}

// fused rmsnorm(over seq) + transpose (+ optional DLinear decompose, K=5 replicate pad)
// X[B][S][C] fp32 -> outA/outT/outS [B][C][S] bf16
template<bool DECOMP>
__global__ __launch_bounds__(256)
void norm_transpose(const float* __restrict__ X, const float* __restrict__ inv,
                    const float* __restrict__ w,
                    bf16* __restrict__ outA, bf16* __restrict__ outT, bf16* __restrict__ outS,
                    int S, int Cc) {
  __shared__ float lds[132 * 33];
  const int b = blockIdx.z;
  const int c0 = blockIdx.y * 32;
  const int s0 = blockIdx.x * 128;
  const int tid = threadIdx.x;
  for (int idx = tid; idx < 132 * 32; idx += 256) {
    int cc = idx & 31;
    int sc = idx >> 5;                       // 0..131 (halo of 2 each side)
    int s = s0 - 2 + sc;
    s = s < 0 ? 0 : (s >= S ? S - 1 : s);    // replicate padding
    lds[sc * 33 + cc] = X[((long)b * S + s) * Cc + c0 + cc] * w[s];
  }
  __syncthreads();
  for (int j = tid; j < 128 * 32; j += 256) {
    int si = j & 127;
    int cc = j >> 7;
    float iv = inv[(long)b * Cc + c0 + cc];
    float av = lds[(si + 2) * 33 + cc] * iv;
    long o = ((long)(b * Cc + c0 + cc)) * (long)S + s0 + si;
    outA[o] = (bf16)av;
    if (DECOMP) {
      float tr = (lds[si * 33 + cc] + lds[(si + 1) * 33 + cc] + lds[(si + 2) * 33 + cc] +
                  lds[(si + 3) * 33 + cc] + lds[(si + 4) * 33 + cc]) * (0.2f * iv);
      outT[o] = (bf16)tr;
      outS[o] = (bf16)(av - tr);
    }
  }
}

// tiled transpose + convert to bf16: in[b][R][Cc] -> out[b][Cc][R]
template<typename InT>
__global__ __launch_bounds__(256)
void transpose_to_bf16(const InT* __restrict__ in, bf16* __restrict__ out,
                       int R, int Cc, long sIn, long sOut) {
  __shared__ float t[32][33];
  const int b = blockIdx.z;
  const int c0 = blockIdx.x * 32;
  const int r0 = blockIdx.y * 32;
  const int tid = threadIdx.x;
  const int tx = tid & 31, ty = tid >> 5;
  for (int j = 0; j < 4; ++j) {
    int r = r0 + ty + 8 * j;
    t[ty + 8 * j][tx] = (float)in[(long)b * sIn + (long)r * Cc + c0 + tx];
  }
  __syncthreads();
  for (int j = 0; j < 4; ++j) {
    int c = c0 + ty + 8 * j;
    out[(long)b * sOut + (long)c * R + r0 + tx] = (bf16)t[tx][ty + 8 * j];
  }
}

// out[b][s][c] = base[b][s][c] + T[b][c][s] * alpha * rowscale[b,c] * colscale[s]
__global__ __launch_bounds__(256)
void transpose_scale_add(const float* __restrict__ base, const float* __restrict__ T,
                         const float* __restrict__ rowscale, const float* __restrict__ colscale,
                         float alpha, float* __restrict__ out, int S, int Cc) {
  __shared__ float t[32][33];
  const int b = blockIdx.z;
  const int s0 = blockIdx.x * 32;
  const int c0 = blockIdx.y * 32;
  const int tid = threadIdx.x;
  const int tx = tid & 31, ty = tid >> 5;
  for (int j = 0; j < 4; ++j) {
    int cc = ty + 8 * j;
    t[cc][tx] = T[((long)(b * Cc + c0 + cc)) * (long)S + s0 + tx];
  }
  __syncthreads();
  for (int j = 0; j < 4; ++j) {
    int s = s0 + ty + 8 * j;
    int c = c0 + tx;
    float f = t[tx][ty + 8 * j] * alpha;
    if (rowscale) f *= rowscale[(long)b * Cc + c];
    if (colscale) f *= colscale[s];
    long o = ((long)b * S + s) * Cc + c;
    out[o] = base[o] + f;
  }
}

// d = softmax(a1) - lam * softmax(a2) per row, written as bf16
__global__ __launch_bounds__(256)
void softmax_combine(const float* __restrict__ A1, const float* __restrict__ A2,
                     const float* __restrict__ lamp, bf16* __restrict__ D, int Cc) {
  __shared__ float red[256];
  const int b = blockIdx.y, i = blockIdx.x, tid = threadIdx.x;
  const long off = ((long)b * Cc + i) * (long)Cc;
  const float* r1 = A1 + off;
  const float* r2 = A2 + off;
  float m1 = -3.4e38f, m2 = -3.4e38f;
  for (int j = tid; j < Cc; j += 256) { m1 = fmaxf(m1, r1[j]); m2 = fmaxf(m2, r2[j]); }
  float M1 = block_reduce(m1, red, tid, true);
  float M2 = block_reduce(m2, red, tid, true);
  float s1 = 0.f, s2 = 0.f;
  for (int j = tid; j < Cc; j += 256) { s1 += expf(r1[j] - M1); s2 += expf(r2[j] - M2); }
  float S1 = block_reduce(s1, red, tid, false);
  float S2 = block_reduce(s2, red, tid, false);
  const float lam = lamp[0];
  const float i1 = 1.f / S1, i2 = lam / S2;
  for (int j = tid; j < Cc; j += 256)
    D[off + j] = (bf16)(expf(r1[j] - M1) * i1 - expf(r2[j] - M2) * i2);
}

// ---------------------------------------------------------------------------
// WMMA GEMM: C[b][M][N] = act( alpha * A[b][M][K] @ Bt[b][N][K]^T + bias[N] )
// A, Bt bf16 row-major; 128x128 block tile, 8 waves of 64x32, K-step 32.
// ---------------------------------------------------------------------------
template<typename OutT, bool GELU>
__global__ __launch_bounds__(256)
void gemm_wmma_bf16(const bf16* __restrict__ A, long lda, long strideA,
                    const bf16* __restrict__ Bt, long ldb, long strideB,
                    const float* __restrict__ bias,
                    OutT* __restrict__ Co, long ldc, long strideC,
                    int K, float alpha) {
  __shared__ bf16 As[128][40];   // row stride 80B: 16B-aligned, bank-friendly
  __shared__ bf16 Bs[128][40];
  const int bz = blockIdx.z;
  const bf16* Ab = A + (long)bz * strideA;
  const bf16* Bb = Bt + (long)bz * strideB;
  const long m0 = (long)blockIdx.y * 128;
  const long n0 = (long)blockIdx.x * 128;
  const int tid = threadIdx.x;
  const int lane = tid & 31;
  const int wave = tid >> 5;
  const int wm = (wave & 1) * 64;   // wave row offset within tile
  const int wn = (wave >> 1) * 32;  // wave col offset within tile
  const int l16 = lane & 15;
  const int lh = lane >> 4;
  const int lrow = tid >> 1;        // 0..127: staging row
  const int lcol = (tid & 1) * 16;  // 0 or 16: staging K-chunk

  v8f acc[4][2];
  for (int i = 0; i < 4; ++i)
    for (int j = 0; j < 2; ++j)
      for (int e = 0; e < 8; ++e) acc[i][j][e] = 0.f;

  const bf16* aptr = Ab + (m0 + lrow) * lda + lcol;
  const bf16* bptr = Bb + (n0 + lrow) * ldb + lcol;

  for (int k0 = 0; k0 < K; k0 += 32) {
    uint4 av = *(const uint4*)(aptr + k0);
    uint4 bv = *(const uint4*)(bptr + k0);
    __builtin_prefetch(aptr + k0 + 32, 0, 1);
    __builtin_prefetch(bptr + k0 + 32, 0, 1);
    *(uint4*)&As[lrow][lcol] = av;
    *(uint4*)&Bs[lrow][lcol] = bv;
    __syncthreads();

    v16bf af[4], bfv[2];
#pragma unroll
    for (int mt = 0; mt < 4; ++mt) {
      // 16-bit A 16x32 layout: lanes 0-15 K{0..7,16..23}, lanes 16-31 K{8..15,24..31}
      const bf16* p = &As[wm + mt * 16 + l16][lh * 8];
      ((uint4*)&af[mt])[0] = *(const uint4*)p;
      ((uint4*)&af[mt])[1] = *(const uint4*)(p + 16);
    }
#pragma unroll
    for (int nt = 0; nt < 2; ++nt) {
      // B 32x16 layout: lanes 0-15 K{0..7,16..23}, lanes 16-31 K{8..15,24..31}
      const bf16* p = &Bs[wn + nt * 16 + l16][lh * 8];
      ((uint4*)&bfv[nt])[0] = *(const uint4*)p;
      ((uint4*)&bfv[nt])[1] = *(const uint4*)(p + 16);
    }
#pragma unroll
    for (int mt = 0; mt < 4; ++mt)
#pragma unroll
      for (int nt = 0; nt < 2; ++nt)
        acc[mt][nt] = __builtin_amdgcn_wmma_f32_16x16x32_bf16(
            false, af[mt], false, bfv[nt], (short)0, acc[mt][nt], false, false);
    __syncthreads();
  }

  // epilogue: D 16x16 f32 layout: VGPR r -> M = r + 8*lh, N = lane&15
#pragma unroll
  for (int mt = 0; mt < 4; ++mt) {
#pragma unroll
    for (int nt = 0; nt < 2; ++nt) {
      long n = n0 + wn + nt * 16 + l16;
      float bvs = bias ? bias[n] : 0.f;
#pragma unroll
      for (int r = 0; r < 8; ++r) {
        long m = m0 + wm + mt * 16 + r + 8 * lh;
        float v = acc[mt][nt][r] * alpha + bvs;
        if (GELU) v = 0.5f * v * (1.f + erff(v * 0.70710678118f));
        Co[(long)bz * strideC + m * ldc + n] = (OutT)v;
      }
    }
  }
}

// ---------------------------------------------------------------------------
// orchestration
// ---------------------------------------------------------------------------
extern "C" void kernel_launch(void* const* d_in, const int* in_sizes, int n_in,
                              void* d_out, int out_size, void* d_ws, size_t ws_size,
                              hipStream_t stream) {
  (void)in_sizes; (void)n_in; (void)out_size; (void)ws_size;
  const int Bn = 16, Sn = 2048, Cn = 512, Hn = 8192;
  const int N2 = 2 * Sn;                 // 4096
  const long BC = (long)Bn * Cn;         // 8192
  const float eps = 1e-8f;
  const float LI = 0.8f - 0.6f * expf(-0.3f);
  const float SCALE = powf((float)Sn, -0.25f);

  const float* x       = (const float*)d_in[0];
  const float* ln_attn = (const float*)d_in[1];
  const float* q12_w   = (const float*)d_in[2];
  const float* q12_b   = (const float*)d_in[3];
  const float* k12_w   = (const float*)d_in[4];
  const float* k12_b   = (const float*)d_in[5];
  const float* v_w     = (const float*)d_in[6];
  const float* v_b     = (const float*)d_in[7];
  const float* lq1     = (const float*)d_in[8];
  const float* lk1     = (const float*)d_in[9];
  const float* lq2     = (const float*)d_in[10];
  const float* lk2     = (const float*)d_in[11];
  const float* norm_w  = (const float*)d_in[12];
  const float* ln_ff   = (const float*)d_in[13];
  const float* ff_w1   = (const float*)d_in[14];
  const float* ff_b1   = (const float*)d_in[15];
  const float* ff_w2   = (const float*)d_in[16];
  const float* ff_b2   = (const float*)d_in[17];
  float* out = (float*)d_out;

  char* wsb = (char*)d_ws;
  size_t cur = 0;
  auto alloc = [&](size_t bytes) -> size_t {
    size_t o = cur; cur += (bytes + 255) & ~(size_t)255; return o;
  };
  size_t off_lam  = alloc(256);
  size_t off_inv  = alloc((size_t)BC * 4);
  size_t off_invn = alloc((size_t)BC * 4);
  size_t off_aT   = alloc((size_t)BC * Sn * 2);   // a^T bf16 (later: h bf16)
  size_t off_trT  = alloc((size_t)BC * Sn * 2);   // trend^T (later: a1,a2 f32)
  size_t off_seT  = alloc((size_t)BC * Sn * 2);   // seasonal^T (later: v^T bf16)
  size_t off_wt   = alloc((size_t)Hn * 2048 * 2); // shared transposed-weight slab
  size_t off_qk   = alloc((size_t)BC * Hn * 2);   // q|k bf16 (later: FFN hidden)
  size_t off_vb   = alloc((size_t)BC * Sn * 2);   // v bf16
  size_t off_db   = alloc((size_t)Bn * Cn * Cn * 2);
  size_t off_dv   = alloc((size_t)BC * Sn * 4);   // d@v f32 (later: ffout f32)

  float* lam  = (float*)(wsb + off_lam);
  float* inv  = (float*)(wsb + off_inv);
  float* invn = (float*)(wsb + off_invn);
  bf16* aT  = (bf16*)(wsb + off_aT);
  bf16* trT = (bf16*)(wsb + off_trT);
  bf16* seT = (bf16*)(wsb + off_seT);
  bf16* wt  = (bf16*)(wsb + off_wt);
  bf16* qb  = (bf16*)(wsb + off_qk);
  bf16* kb  = qb + (size_t)BC * N2;
  bf16* gb  = qb;                            // FFN hidden reuses q|k (128 MiB)
  bf16* vb  = (bf16*)(wsb + off_vb);
  bf16* vTb = (bf16*)(wsb + off_seT);        // reuse seasonal slab
  float* a1 = (float*)(wsb + off_trT);       // reuse trend slab
  float* a2 = a1 + (size_t)Bn * Cn * Cn;
  bf16* db  = (bf16*)(wsb + off_db);
  float* dv = (float*)(wsb + off_dv);
  bf16* hb  = (bf16*)(wsb + off_aT);         // reuse a^T slab
  float* ffout = (float*)(wsb + off_dv);     // reuse d@v slab

  // 1) lambda scalar
  lam_kernel<<<1, 256, 0, stream>>>(lq1, lk1, lq2, lk2, lam, Sn, LI);
  // 2) rms over seq of x; 3) fused norm+decompose+transpose -> aT/trT/seT (bf16)
  colstats<<<dim3(Cn / 32, Bn), 256, 0, stream>>>(x, inv, Sn, Cn, eps);
  norm_transpose<true><<<dim3(Sn / 128, Cn / 32, Bn), 256, 0, stream>>>(
      x, inv, ln_attn, aT, trT, seT, Sn, Cn);
  // 4) q = trend^T @ q12_w + b  (weights converted+transposed to bf16 [N][K])
  transpose_to_bf16<float><<<dim3(N2 / 32, Sn / 32, 1), 256, 0, stream>>>(q12_w, wt, Sn, N2, 0, 0);
  gemm_wmma_bf16<bf16, false><<<dim3(N2 / 128, BC / 128, 1), 256, 0, stream>>>(
      trT, Sn, 0, wt, Sn, 0, q12_b, qb, N2, 0, Sn, 1.f);
  // 5) k = seasonal^T @ k12_w + b
  transpose_to_bf16<float><<<dim3(N2 / 32, Sn / 32, 1), 256, 0, stream>>>(k12_w, wt, Sn, N2, 0, 0);
  gemm_wmma_bf16<bf16, false><<<dim3(N2 / 128, BC / 128, 1), 256, 0, stream>>>(
      seT, Sn, 0, wt, Sn, 0, k12_b, kb, N2, 0, Sn, 1.f);
  // 6) v = a^T @ v_w + b
  transpose_to_bf16<float><<<dim3(Sn / 32, Sn / 32, 1), 256, 0, stream>>>(v_w, wt, Sn, Sn, 0, 0);
  gemm_wmma_bf16<bf16, false><<<dim3(Sn / 128, BC / 128, 1), 256, 0, stream>>>(
      aT, Sn, 0, wt, Sn, 0, v_b, vb, Sn, 0, Sn, 1.f);
  // 7) v^T per batch (for d@v as A@Bt^T)
  transpose_to_bf16<bf16><<<dim3(Sn / 32, Cn / 32, Bn), 256, 0, stream>>>(
      vb, vTb, Cn, Sn, (long)Cn * Sn, (long)Cn * Sn);
  // 8) scores: a1 = SCALE * q1 @ k1^T, a2 = SCALE * q2 @ k2^T (batched)
  gemm_wmma_bf16<float, false><<<dim3(Cn / 128, Cn / 128, Bn), 256, 0, stream>>>(
      qb, N2, (long)Cn * N2, kb, N2, (long)Cn * N2, nullptr, a1, Cn, (long)Cn * Cn, Sn, SCALE);
  gemm_wmma_bf16<float, false><<<dim3(Cn / 128, Cn / 128, Bn), 256, 0, stream>>>(
      qb + Sn, N2, (long)Cn * N2, kb + Sn, N2, (long)Cn * N2, nullptr, a2, Cn, (long)Cn * Cn, Sn, SCALE);
  // 9) d = softmax(a1) - lam*softmax(a2)
  softmax_combine<<<dim3(Cn, Bn), 256, 0, stream>>>(a1, a2, lam, db, Cn);
  // 10) dv = d @ v  (batched, K = C)
  gemm_wmma_bf16<float, false><<<dim3(Sn / 128, Cn / 128, Bn), 256, 0, stream>>>(
      db, Cn, (long)Cn * Cn, vTb, Cn, (long)Sn * Cn, nullptr, dv, Sn, (long)Cn * Sn, Cn, 1.f);
  // 11) rmsnorm(dv) scale + 12) out = x + attn^T
  rowstats<<<dim3((unsigned)BC), 256, 0, stream>>>(dv, invn, Sn, eps);
  transpose_scale_add<<<dim3(Sn / 32, Cn / 32, Bn), 256, 0, stream>>>(
      x, dv, invn, norm_w, 1.f - LI, out, Sn, Cn);
  // 13) FFN pre-norm over seq; 14) h bf16 [B*C][S]
  colstats<<<dim3(Cn / 32, Bn), 256, 0, stream>>>(out, inv, Sn, Cn, eps);
  norm_transpose<false><<<dim3(Sn / 128, Cn / 32, Bn), 256, 0, stream>>>(
      out, inv, ln_ff, hb, nullptr, nullptr, Sn, Cn);
  // 15) g = gelu(h @ ff_w1 + b1) bf16
  transpose_to_bf16<float><<<dim3(Hn / 32, Sn / 32, 1), 256, 0, stream>>>(ff_w1, wt, Sn, Hn, 0, 0);
  gemm_wmma_bf16<bf16, true><<<dim3(Hn / 128, BC / 128, 1), 256, 0, stream>>>(
      hb, Sn, 0, wt, Sn, 0, ff_b1, gb, Hn, 0, Sn, 1.f);
  // 16) ff = g @ ff_w2 + b2 (f32)
  transpose_to_bf16<float><<<dim3(Sn / 32, Hn / 32, 1), 256, 0, stream>>>(ff_w2, wt, Hn, Sn, 0, 0);
  gemm_wmma_bf16<float, false><<<dim3(Sn / 128, BC / 128, 1), 256, 0, stream>>>(
      gb, Hn, 0, wt, Hn, 0, ff_b2, ffout, Sn, 0, Hn, 1.f);
  // 17) out += ff^T (in place)
  transpose_scale_add<<<dim3(Sn / 32, Cn / 32, Bn), 256, 0, stream>>>(
      out, ffout, nullptr, nullptr, 1.f, out, Sn, Cn);
}
